// PraxisMixtureOfDepths_14577119002816
// MI455X (gfx1250) — matrix operand order repaired and checked by reading, not compiled, not run on hardware
//
#include <hip/hip_runtime.h>

typedef unsigned short u16;
typedef unsigned long long u64;
typedef __attribute__((ext_vector_type(8)))  __bf16 v8bf;
typedef __attribute__((ext_vector_type(16))) __bf16 v16bf;
typedef __attribute__((ext_vector_type(8)))  float  v8f;

#define BATCH  4
#define SEQ    4096
#define DMODEL 1024
#define DFF    4096
#define KSEL   2048            // SEQ * CAPACITY(0.5)
#define MROWS  (BATCH * KSEL)  // 8192 selected rows total

// ---------------- helpers ----------------

__device__ __forceinline__ u16 f32_to_bf16(float f) {
  unsigned u = __float_as_uint(f);
  u += 0x7FFFu + ((u >> 16) & 1u);   // round to nearest even
  return (u16)(u >> 16);
}

__device__ __forceinline__ float wave_sum(float v) {
#pragma unroll
  for (int off = 16; off > 0; off >>= 1) v += __shfl_xor(v, off, 32);
  return v;
}

__device__ __forceinline__ float gelu_tanh(float x) {
  float x3 = x * x * x;
  return 0.5f * x * (1.0f + tanhf(0.7978845608028654f * (x + 0.044715f * x3)));
}

__device__ __forceinline__ v8f wmma_bf16(v16bf a, v16bf b, v8f c) {
  // v_wmma_f32_16x16x32_bf16: (neg_a, A, neg_b, B, c_mod, C, reuse_a, reuse_b)
  return __builtin_amdgcn_wmma_f32_16x16x32_bf16(false, a, false, b, (short)0, c,
                                                 false, false);
}

// A fragment (16x32 bf16, MxK): lane<16 holds row M=lane%16, K = {g..g+7,
// g+16..g+23} with g = (lane<16 ? 0 : 8). Two contiguous 16B chunks.
__device__ __forceinline__ v16bf load_a_frag(const u16* p) {
  v8bf lo = *(const v8bf*)(p);
  v8bf hi = *(const v8bf*)(p + 16);
  return __builtin_shufflevector(lo, hi, 0, 1, 2, 3, 4, 5, 6, 7, 8, 9, 10, 11, 12,
                                 13, 14, 15);
}

// B fragment (32x16 bf16, KxN): lane L holds column N=L%16, K = {g..g+7,
// g+16..g+23} with g = (L<16 ? 0 : 8). Two contiguous 16B chunks.
__device__ __forceinline__ v16bf load_b_frag(const u16* p) {
  v8bf lo = *(const v8bf*)(p);
  v8bf hi = *(const v8bf*)(p + 16);
  return __builtin_shufflevector(lo, hi, 0, 1, 2, 3, 4, 5, 6, 7, 8, 9, 10, 11, 12,
                                 13, 14, 15);
}

// Async global->LDS copy of 16 bytes per lane (CDNA5, tracked by ASYNCcnt).
// VDST holds the LDS byte address; for a generic pointer to __shared__ memory
// the low 32 bits ARE the LDS offset (ISA 10.2 aperture rule).
__device__ __forceinline__ void async_copy16(const u16* gsrc, const u16* lds_dst) {
  unsigned lds_off = (unsigned)(size_t)lds_dst;
  asm volatile("global_load_async_to_lds_b128 %0, %1, off"
               :: "v"(lds_off), "v"(gsrc)
               : "memory");
}

__device__ __forceinline__ void wait_async0() {
  asm volatile("s_wait_asynccnt 0" ::: "memory");
}

// ---------------- shared GEMM core ----------------
// Block = 256 thr = 8 waves -> 128(M) x 64(N) tile; each wave 32x32 via 2x2 WMMA.
// Double-buffered async-LDS pipeline over K in steps of 32.
__device__ __forceinline__ void gemm_core(const u16* __restrict__ Agl,
                                          const u16* __restrict__ BTgl, int K,
                                          int m0blk, int n0blk, u16* sA, u16* sB,
                                          v8f acc[2][2]) {
  const int tid = threadIdx.x;
  const int lane = tid & 31;
  const int wid = tid >> 5;
  const int waveM = wid & 3;
  const int waveN = wid >> 2;
  const int lm = lane & 15;
  const int kgA = (lane < 16) ? 0 : 8;
  const int kgB = (lane < 16) ? 0 : 8;

  // Stage one 128x32 A tile + 64x32 B tile (bf16) into LDS buffer `buf`.
  auto stage = [&](int buf, int kt) {
    {  // B: 64 rows x 64B = 256 16B chunks, one per thread
      int r = tid >> 2, c = tid & 3;
      const u16* src = BTgl + (size_t)(n0blk + r) * K + kt + c * 8;
      async_copy16(src, sB + buf * (64 * 32) + r * 32 + c * 8);
    }
#pragma unroll
    for (int i = 0; i < 2; i++) {  // A: 128 rows x 64B = 512 chunks, two/thread
      int idx = tid + i * 256;
      int r = idx >> 2, c = idx & 3;
      const u16* src = Agl + (size_t)(m0blk + r) * K + kt + c * 8;
      if (i == 0) __builtin_prefetch(src + 64, 0, 0);  // next-next K tile
      async_copy16(src, sA + buf * (128 * 32) + r * 32 + c * 8);
    }
  };

  stage(0, 0);
  wait_async0();
  __syncthreads();

  for (int kt = 0; kt < K; kt += 32) {
    int buf = (kt >> 5) & 1;
    if (kt + 32 < K) stage(buf ^ 1, kt + 32);  // overlap copy with compute

    const u16* a0p = sA + buf * (128 * 32) + (waveM * 32 + lm) * 32 + kgA;
    const u16* a1p = a0p + 16 * 32;
    const u16* b0p = sB + buf * (64 * 32) + (waveN * 32 + lm) * 32 + kgB;
    const u16* b1p = b0p + 16 * 32;
    v16bf a0 = load_a_frag(a0p);
    v16bf a1 = load_a_frag(a1p);
    v16bf b0 = load_b_frag(b0p);
    v16bf b1 = load_b_frag(b1p);
    acc[0][0] = wmma_bf16(a0, b0, acc[0][0]);
    acc[0][1] = wmma_bf16(a0, b1, acc[0][1]);
    acc[1][0] = wmma_bf16(a1, b0, acc[1][0]);
    acc[1][1] = wmma_bf16(a1, b1, acc[1][1]);

    if (kt + 32 < K) {
      wait_async0();     // own wave's async copies done
      __syncthreads();   // all waves' copies visible; readers of old buf done
    }
  }
}

// ---------------- kernels ----------------

// One wave per token: logit = x[token,:]·Wr + br, score = sigmoid(logit).
__global__ __launch_bounds__(256) void router_kernel(const float* __restrict__ x,
                                                     const float* __restrict__ Wr,
                                                     const float* __restrict__ br,
                                                     float* __restrict__ logits,
                                                     float* __restrict__ scores) {
  int token = blockIdx.x * 8 + (threadIdx.x >> 5);
  int lane = threadIdx.x & 31;
  const float* xr = x + (size_t)token * DMODEL;
  float s = 0.0f;
#pragma unroll 8
  for (int i = 0; i < DMODEL / 32; i++) s += xr[i * 32 + lane] * Wr[i * 32 + lane];
  s = wave_sum(s);
  if (lane == 0) {
    float l = s + br[0];
    logits[token] = l;
    scores[token] = 1.0f / (1.0f + expf(-l));
  }
}

__global__ void zero_loss_kernel(float* __restrict__ loss) { *loss = 0.0f; }

// One block per batch. Exact top-k via bitonic sort of (score_bits, ~idx) keys,
// softmax over selected scores, ascending-index compaction via prefix scan, and
// the BCE aux-loss reduction.
__global__ __launch_bounds__(1024) void select_kernel(
    const float* __restrict__ scores, const float* __restrict__ logits,
    int* __restrict__ sel_idx, float* __restrict__ rw_out,
    float* __restrict__ loss) {
  __shared__ u64 keys[SEQ];        // 32 KB
  __shared__ float rwl[SEQ];       // 16 KB
  __shared__ unsigned scan[1024];  // 4 KB
  __shared__ float wred[32];
  __shared__ float s_total;

  const int batch = blockIdx.x;
  const int tid = threadIdx.x;
  const int nthr = 1024;
  const int lane = tid & 31;
  const int wid = tid >> 5;

  // Pack keys: sigmoid scores are positive -> float bits are order-preserving.
  // Low 32 bits = (SEQ-1 - idx) so descending sort breaks ties by LOWER index.
  for (int i = tid; i < SEQ; i += nthr) {
    unsigned bits = __float_as_uint(scores[batch * SEQ + i]);
    keys[i] = ((u64)bits << 32) | (u64)(unsigned)(SEQ - 1 - i);
  }
  __syncthreads();

  // Bitonic sort, descending (largest key first).
  for (unsigned k = 2; k <= SEQ; k <<= 1) {
    for (unsigned j = k >> 1; j > 0; j >>= 1) {
      for (unsigned i = tid; i < SEQ; i += nthr) {
        unsigned ixj = i ^ j;
        if (ixj > i) {
          u64 a = keys[i], b = keys[ixj];
          bool desc = ((i & k) == 0);
          if (desc ? (a < b) : (a > b)) {
            keys[i] = b;
            keys[ixj] = a;
          }
        }
      }
      __syncthreads();
    }
  }

  // Softmax over the top-KSEL scores (permutation invariant).
  float m = __uint_as_float((unsigned)(keys[0] >> 32));
  float acc = 0.0f;
  for (int e = tid; e < KSEL; e += nthr)
    acc += expf(__uint_as_float((unsigned)(keys[e] >> 32)) - m);
  acc = wave_sum(acc);
  if (lane == 0) wred[wid] = acc;
  __syncthreads();
  if (tid < 32) {
    float t = wred[tid];
    t = wave_sum(t);
    if (tid == 0) s_total = t;
  }
  __syncthreads();
  float total = s_total;

  // Per-token router weight; -1 sentinel = not selected.
  for (int i = tid; i < SEQ; i += nthr) rwl[i] = -1.0f;
  __syncthreads();
  for (int e = tid; e < KSEL; e += nthr) {
    u64 key = keys[e];
    unsigned idx = (SEQ - 1) - (unsigned)(key & 0xFFFFFFFFull);
    float sc = __uint_as_float((unsigned)(key >> 32));
    rwl[idx] = expf(sc - m) / total;
  }
  __syncthreads();

  // Compact selected tokens in ascending index order (== sorted_indices).
  unsigned cnt = 0;
#pragma unroll
  for (int t = 0; t < 4; t++) cnt += (rwl[tid * 4 + t] >= 0.0f) ? 1u : 0u;
  scan[tid] = cnt;
  __syncthreads();
  for (int off = 1; off < 1024; off <<= 1) {
    unsigned v = scan[tid];
    unsigned a = (tid >= off) ? scan[tid - off] : 0u;
    __syncthreads();
    scan[tid] = v + a;
    __syncthreads();
  }
  unsigned base = scan[tid] - cnt;
#pragma unroll
  for (int t = 0; t < 4; t++) {
    int tok = tid * 4 + t;
    if (rwl[tok] >= 0.0f) {
      sel_idx[batch * KSEL + base] = tok;
      rw_out[batch * KSEL + base] = rwl[tok];
      base++;
    }
  }

  // Aux loss: mean over (b,s) of max(l,0) - l*target + log1p(exp(-|l|)).
  float ls = 0.0f;
  for (int i = tid; i < SEQ; i += nthr) {
    float l = logits[batch * SEQ + i];
    float tgt = (rwl[i] >= 0.0f) ? 1.0f : 0.0f;
    ls += fmaxf(l, 0.0f) - l * tgt + log1pf(expf(-fabsf(l)));
  }
  ls = wave_sum(ls);
  if (lane == 0) atomicAdd(loss, ls * (1.0f / (float)(BATCH * SEQ)));
}

// W1 [d, dff] f32 -> W1T [dff, d] bf16 (K-contiguous rows for B fragments).
__global__ __launch_bounds__(256) void cast_w1t_kernel(const float* __restrict__ W1,
                                                       u16* __restrict__ W1T) {
  int t = blockIdx.x * 256 + threadIdx.x;
  int k = t / DFF, f = t % DFF;
  W1T[(size_t)f * DMODEL + k] = f32_to_bf16(W1[t]);
}

// W2 [dff, d] f32 -> W2T [d, dff] bf16.
__global__ __launch_bounds__(256) void cast_w2t_kernel(const float* __restrict__ W2,
                                                       u16* __restrict__ W2T) {
  int t = blockIdx.x * 256 + threadIdx.x;
  int f = t / DMODEL, n = t % DMODEL;
  W2T[(size_t)n * DFF + f] = f32_to_bf16(W2[t]);
}

// Gather selected rows of x into the dense bf16 A matrix [MROWS, d].
__global__ __launch_bounds__(256) void gather_cast_kernel(
    const float* __restrict__ x, const int* __restrict__ sel_idx,
    u16* __restrict__ A) {
  int r = blockIdx.x;  // 0..MROWS-1
  int bidx = r >> 11;  // r / KSEL
  int token = sel_idx[r];
  const float* src = x + ((size_t)(bidx * SEQ + token)) * DMODEL;
  u16* dst = A + (size_t)r * DMODEL;
  for (int i = threadIdx.x; i < DMODEL; i += 256) dst[i] = f32_to_bf16(src[i]);
}

// Pass-through: hidden_states starts as a copy of x.
__global__ __launch_bounds__(256) void copy_x_kernel(const float4* __restrict__ x,
                                                     float4* __restrict__ out) {
  size_t i = (size_t)blockIdx.x * 256 + threadIdx.x;
  out[i] = x[i];
}

// GEMM1: H = gelu(A[8192,1024] x W1[1024,4096] + b1), stored bf16.
__global__ __launch_bounds__(256) void gemm1_kernel(const u16* __restrict__ A,
                                                    const u16* __restrict__ BT,
                                                    const float* __restrict__ bias,
                                                    u16* __restrict__ H) {
  __shared__ __align__(16) u16 sA[2 * 128 * 32];
  __shared__ __align__(16) u16 sB[2 * 64 * 32];
  const int K = DMODEL, N = DFF;
  int wid = threadIdx.x >> 5;
  int lane = threadIdx.x & 31;
  int m0blk = blockIdx.y * 128;
  int n0blk = blockIdx.x * 64;
  int m0 = m0blk + (wid & 3) * 32;
  int n0 = n0blk + (wid >> 2) * 32;

  v8f acc[2][2] = {{{}, {}}, {{}, {}}};
  gemm_core(A, BT, K, m0blk, n0blk, sA, sB, acc);

  // C/D layout: VGPR j -> M = j (lanes 0-15) or j+8 (lanes 16-31); N = lane%16.
  int nn = lane & 15;
  int mb = (lane < 16) ? 0 : 8;
#pragma unroll
  for (int tm = 0; tm < 2; tm++)
#pragma unroll
    for (int tn = 0; tn < 2; tn++)
#pragma unroll
      for (int j = 0; j < 8; j++) {
        int row = m0 + tm * 16 + mb + j;
        int col = n0 + tn * 16 + nn;
        float v = acc[tm][tn][j] + bias[col];
        H[(size_t)row * N + col] = f32_to_bf16(gelu_tanh(v));
      }
}

// GEMM2: out_rows = H[8192,4096] x W2[4096,1024] + b2, fused scatter epilogue:
// d_out[b, token, :] = x[b, token, :] + rw * out_row.
__global__ __launch_bounds__(256) void gemm2_kernel(
    const u16* __restrict__ H, const u16* __restrict__ BT,
    const float* __restrict__ bias, const int* __restrict__ sel_idx,
    const float* __restrict__ rw, const float* __restrict__ x,
    float* __restrict__ out) {
  __shared__ __align__(16) u16 sA[2 * 128 * 32];
  __shared__ __align__(16) u16 sB[2 * 64 * 32];
  const int K = DFF;
  int wid = threadIdx.x >> 5;
  int lane = threadIdx.x & 31;
  int m0blk = blockIdx.y * 128;
  int n0blk = blockIdx.x * 64;
  int m0 = m0blk + (wid & 3) * 32;
  int n0 = n0blk + (wid >> 2) * 32;

  v8f acc[2][2] = {{{}, {}}, {{}, {}}};
  gemm_core(H, BT, K, m0blk, n0blk, sA, sB, acc);

  int nn = lane & 15;
  int mb = (lane < 16) ? 0 : 8;
#pragma unroll
  for (int tm = 0; tm < 2; tm++)
#pragma unroll
    for (int tn = 0; tn < 2; tn++)
#pragma unroll
      for (int j = 0; j < 8; j++) {
        int r = m0 + tm * 16 + mb + j;  // selected-row index in [0, MROWS)
        int bidx = r >> 11;
        int token = sel_idx[r];
        float w = rw[r];
        int col = n0 + tn * 16 + nn;
        float v = acc[tm][tn][j] + bias[col];
        size_t gi = ((size_t)(bidx * SEQ + token)) * DMODEL + col;
        out[gi] = x[gi] + w * v;
      }
}

// ---------------- host launcher ----------------

extern "C" void kernel_launch(void* const* d_in, const int* in_sizes, int n_in,
                              void* d_out, int out_size, void* d_ws, size_t ws_size,
                              hipStream_t stream) {
  const float* x = (const float*)d_in[0];
  // d_in[1] = attention_mask (unused by the reference computation)
  const float* Wr = (const float*)d_in[2];
  const float* br = (const float*)d_in[3];
  const float* W1 = (const float*)d_in[4];
  const float* b1 = (const float*)d_in[5];
  const float* W2 = (const float*)d_in[6];
  const float* b2 = (const float*)d_in[7];
  float* out = (float*)d_out;
  float* loss = out + (size_t)BATCH * SEQ * DMODEL;

  char* ws = (char*)d_ws;
  float* scores = (float*)ws; ws += (size_t)BATCH * SEQ * 4;
  float* logits = (float*)ws; ws += (size_t)BATCH * SEQ * 4;
  int* sel      = (int*)ws;   ws += (size_t)MROWS * 4;
  float* rw     = (float*)ws; ws += (size_t)MROWS * 4;
  u16* A   = (u16*)ws; ws += (size_t)MROWS * DMODEL * 2;
  u16* W1T = (u16*)ws; ws += (size_t)DFF * DMODEL * 2;
  u16* W2T = (u16*)ws; ws += (size_t)DMODEL * DFF * 2;
  u16* Hbuf = (u16*)ws; ws += (size_t)MROWS * DFF * 2;

  router_kernel<<<(BATCH * SEQ) / 8, 256, 0, stream>>>(x, Wr, br, logits, scores);
  zero_loss_kernel<<<1, 1, 0, stream>>>(loss);
  select_kernel<<<BATCH, 1024, 0, stream>>>(scores, logits, sel, rw, loss);
  cast_w1t_kernel<<<(DMODEL * DFF) / 256, 256, 0, stream>>>(W1, W1T);
  cast_w2t_kernel<<<(DFF * DMODEL) / 256, 256, 0, stream>>>(W2, W2T);
  gather_cast_kernel<<<MROWS, 256, 0, stream>>>(x, sel, A);
  copy_x_kernel<<<(BATCH * SEQ * DMODEL / 4) / 256, 256, 0, stream>>>(
      (const float4*)x, (float4*)out);
  gemm1_kernel<<<dim3(DFF / 64, MROWS / 128), 256, 0, stream>>>(A, W1T, b1, Hbuf);
  gemm2_kernel<<<dim3(DMODEL / 64, MROWS / 128), 256, 0, stream>>>(Hbuf, W2T, b2,
                                                                   sel, rw, x, out);
}
